// AxialAttention_77034533421570
// MI455X (gfx1250) — matrix operand order repaired and checked by reading, not compiled, not run on hardware
//
#include <hip/hip_runtime.h>

// Axial attention (height axis) for MI455X / gfx1250, wave32 + WMMA f16.
// qkv_proj (WMMA GEMM) -> per-(seq,head) fused attention (WMMA) -> out_proj
// (WMMA GEMM). f16 intermediates in d_ws (~268 MB needed).
//
// Q/K/V are stored in GLOBAL memory as WMMA *fragment images* per
// (bt,head,w) block (8 tiles x 32 lanes x 16 halves = 8KB): the attention
// kernel loads operand fragments with plain coalesced 16B reads (global or
// LDS) -- no per-element gathering anywhere on the WMMA read path.

typedef _Float16 half_t;
typedef __attribute__((ext_vector_type(16))) _Float16 v16h;
typedef __attribute__((ext_vector_type(2)))  _Float16 v2h;
typedef __attribute__((ext_vector_type(2)))  __fp16   v2fp16;
typedef __attribute__((ext_vector_type(8)))  float    v8f;

#define DIMC    256
#define NHEADS  8
#define DH      32
#define LSEQ    128   // H
#define WDIM    128   // W
#define BT      8     // B*T
#define THREE_C 768
#define SCALE_F 0.17677669529663687f  // 1/sqrt(32)

// A-frag element (m,k): lane=(m&15)+16*(k>>4), e=k&15
__device__ __forceinline__ int a_off(int m, int k) {
  return (((m & 15) + ((k >> 4) << 4)) << 4) | (k & 15);
}
// B-frag element (k,n): lane=(n&15)+16*(k>>4), e=k&15
__device__ __forceinline__ int b_off(int k, int n) {
  return (((n & 15) + ((k >> 4) << 4)) << 4) | (k & 15);
}

__device__ __forceinline__ v2h pk2(float a, float b) {
#if __has_builtin(__builtin_amdgcn_cvt_pkrtz)
  union { v2fp16 f; v2h h; } cv;
  cv.f = __builtin_amdgcn_cvt_pkrtz(a, b);  // v_cvt_pk_rtz_f16_f32
  return cv.h;
#else
  v2h r; r[0] = (half_t)a; r[1] = (half_t)b; return r;
#endif
}

union Pack8 { v2h h2[4]; uint4 u; };

#define WMMA16(a, b, c) \
  __builtin_amdgcn_wmma_f32_16x16x32_f16(false, (a), false, (b), (short)0, (c), false, false)

// ---------------- Kernel 1: qkv = seq @ Wqkv + bqkv ----------------
// Block = one (bt,l) x 64 output cols. A[w][c] = x[bt,c,l,w] (contiguous in w).
__global__ __launch_bounds__(256) void qkv_proj_kernel(
    const float* __restrict__ x, const float* __restrict__ Wqkv,
    const float* __restrict__ bqkv, half_t* __restrict__ Q,
    half_t* __restrict__ K, half_t* __restrict__ V) {
  __shared__ v16h a_fr[8 * 32];  // 8 m-tiles, frag order
  __shared__ v16h b_fr[4 * 32];  // 4 n-tiles, frag order
  half_t* a_h = (half_t*)a_fr;
  half_t* b_h = (half_t*)b_fr;

  const int bx = blockIdx.x;
  const int bt = bx >> 7, l = bx & 127;
  const int j0 = blockIdx.y * 64;
  const int tid = threadIdx.x;
  const int lane = tid & 31, wid = tid >> 5;
  const int wm = wid & 3, wn = wid >> 2;  // 4 m-groups x 2 n-groups

  v8f acc[2][2] = {};

  for (int ks = 0; ks < 8; ++ks) {
    const int k0 = ks * 32;
    __syncthreads();
    {  // A tile: thread owns 16 strided c at fixed w; lane-coalesced b32 loads,
       // two packed 16B frag-order LDS stores
      const int w = tid & 127;
      const int kh = (tid >> 7) * 16;  // k base 0 / 16
      const float* src =
          x + (((size_t)(bt * DIMC + k0 + kh) * LSEQ + l) * WDIM) + w;
      const size_t cs = (size_t)LSEQ * WDIM;
      Pack8 lo, hi;
#pragma unroll
      for (int u = 0; u < 4; ++u) {
        lo.h2[u] = pk2(src[(2 * u) * cs], src[(2 * u + 1) * cs]);
        hi.h2[u] = pk2(src[(8 + 2 * u) * cs], src[(9 + 2 * u) * cs]);
      }
      *(uint4*)(a_h + (w >> 4) * 512 + a_off(w, kh)) = lo.u;
      *(uint4*)(a_h + (w >> 4) * 512 + a_off(w, kh + 8)) = hi.u;
    }
    {  // B tile: thread owns 8 strided c at fixed j; one packed 16B store
      const int nn = tid & 63;
      const int kg = tid >> 6;  // 0..3
      const float* src = Wqkv + (size_t)(k0 + kg * 8) * THREE_C + j0 + nn;
      Pack8 pk;
#pragma unroll
      for (int u = 0; u < 4; ++u)
        pk.h2[u] = pk2(src[(2 * u) * THREE_C], src[(2 * u + 1) * THREE_C]);
      *(uint4*)(b_h + (nn >> 4) * 512 + b_off(kg * 8, nn)) = pk.u;
    }
    __syncthreads();
#pragma unroll
    for (int im = 0; im < 2; ++im) {
      const v16h a = a_fr[(wm * 2 + im) * 32 + lane];
#pragma unroll
      for (int in_ = 0; in_ < 2; ++in_) {
        const v16h b = b_fr[(wn * 2 + in_) * 32 + lane];
        acc[im][in_] = WMMA16(a, b, acc[im][in_]);
      }
    }
  }

  // Epilogue: +bqkv; write Q/K/V fragment images [bt][head][w] blocks of 4096.
  // Within-block offset depends only on (l,dh) -> constant per lane.
#pragma unroll
  for (int in_ = 0; in_ < 2; ++in_) {
    const int nt = wn * 2 + in_;
    const int j = j0 + nt * 16 + (lane & 15);
    const int qsel = j >> 8;  // uniform across the 16-col tile
    const int jj = j & 255;
    const int head = jj >> 5, dh = jj & 31;
    const float bj = bqkv[j];
    half_t* dstp;
    int off_in;
    if (qsel == 0) {        // Q: A-frag image (m=l, k=dh), tile l>>4
      dstp = Q; off_in = (l >> 4) * 512 + a_off(l, dh);
    } else if (qsel == 1) { // K: B-frag image for K^T (k=dh, n=l), tile l>>4
      dstp = K; off_in = (l >> 4) * 512 + b_off(dh, l);
    } else {                // V: B-frag image (k=l, n=dh), tile (l>>5)*2+(dh>>4)
      dstp = V; off_in = ((l >> 5) * 2 + (dh >> 4)) * 512 + b_off(l & 31, dh);
    }
    const size_t hb = (size_t)(bt * NHEADS + head) * WDIM;
#pragma unroll
    for (int im = 0; im < 2; ++im) {
      const int mt = wm * 2 + im;
#pragma unroll
      for (int r = 0; r < 8; ++r) {
        const int w = mt * 16 + r + 8 * (lane >> 4);
        dstp[(hb + w) * 4096 + off_in] = (half_t)(acc[im][in_][r] + bj);
      }
    }
  }
}

// ---------------- Kernel 2: fused attention per (n, head) ----------------
// 2 waves/block; each wave owns one (n,head). Q fragments read straight from
// global (single use), K fragments hoisted to registers (8x reuse), V image
// staged to LDS with plain 16B copies, P transposed via LDS scatter.
__global__ __launch_bounds__(64) void attn_kernel(
    const half_t* __restrict__ Qws, const half_t* __restrict__ Kws,
    const half_t* __restrict__ Vws, const float* __restrict__ bias,
    half_t* __restrict__ O) {
  __shared__ v16h smem2[2][12 * 32];  // per wave: V 8 tiles + P 4 tiles (12KB)

  const int tid = threadIdx.x;
  const int lane = tid & 31, wid = tid >> 5;
  const int gid = blockIdx.x * 2 + wid;
  const int head = gid & 7;
  const int n = gid >> 3;
  const int bt = n >> 7, w = n & 127;

  v16h* v_fr = smem2[wid];
  v16h* p_fr = v_fr + 8 * 32;
  half_t* p_h = (half_t*)p_fr;

  const size_t base = (size_t)((bt * NHEADS + head) * WDIM + w) * 4096;
  const v16h* qg = (const v16h*)(Qws + base);

  {  // stage V fragment image: plain coalesced 16B copies
    const uint4* vs = (const uint4*)(Vws + base);
    uint4* vd = (uint4*)v_fr;
#pragma unroll
    for (int i = 0; i < 16; ++i) vd[i * 32 + lane] = vs[i * 32 + lane];
  }
  // hoist K fragments (each reused by all 8 row blocks)
  v16h kfr[8];
  {
    const v16h* kg = (const v16h*)(Kws + base);
#pragma unroll
    for (int lt = 0; lt < 8; ++lt) kfr[lt] = kg[lt * 32 + lane];
  }
  __syncthreads();

  const float* brow = bias + (size_t)head * LSEQ * LSEQ;
  const int ncol = lane & 15, hsel = lane >> 4;

  for (int mt = 0; mt < 8; ++mt) {
    if (mt < 7)  // prefetch next bias row block (global_prefetch_b8)
      __builtin_prefetch(brow + (size_t)(mt + 1) * 16 * LSEQ + lane * 16, 0, 0);
    // ---- S row-block: 16 x 128 ----
    const v16h qa = qg[mt * 32 + lane];  // direct global fragment load
    v8f s[8];
#pragma unroll
    for (int lt = 0; lt < 8; ++lt) {
      const v8f z = {};
      s[lt] = WMMA16(qa, kfr[lt], z);
    }
    // ---- scale + bias + row softmax (8 tiles x 16 lanes per row) ----
#pragma unroll
    for (int r = 0; r < 8; ++r) {
      const int m = r + 8 * hsel;  // D-layout row within tile
      const int lr = mt * 16 + m;
      float mx = -3.0e38f;
#pragma unroll
      for (int lt = 0; lt < 8; ++lt) {
        const float val = s[lt][r] * SCALE_F + brow[lr * LSEQ + lt * 16 + ncol];
        s[lt][r] = val;
        mx = fmaxf(mx, val);
      }
      mx = fmaxf(mx, __shfl_xor(mx, 1, 32));
      mx = fmaxf(mx, __shfl_xor(mx, 2, 32));
      mx = fmaxf(mx, __shfl_xor(mx, 4, 32));
      mx = fmaxf(mx, __shfl_xor(mx, 8, 32));
      float sum = 0.f;
#pragma unroll
      for (int lt = 0; lt < 8; ++lt) {
        const float e = __expf(s[lt][r] - mx);
        s[lt][r] = e;
        sum += e;
      }
      sum += __shfl_xor(sum, 1, 32);
      sum += __shfl_xor(sum, 2, 32);
      sum += __shfl_xor(sum, 4, 32);
      sum += __shfl_xor(sum, 8, 32);
      const float inv = 1.0f / sum;
#pragma unroll
      for (int lt = 0; lt < 8; ++lt) {  // write P in A-frag order
        const int col = lt * 16 + ncol;
        p_h[(col >> 5) * 512 + a_off(m, col & 31)] = (half_t)(s[lt][r] * inv);
      }
    }
    // same-wave DS ordering (ISA 7.3) makes p_fr visible to the reads below
    // ---- O row-block: (16x128) @ (128x32) ----
    v8f o[2] = {v8f{}, v8f{}};
#pragma unroll
    for (int kt = 0; kt < 4; ++kt) {
      const v16h pa = p_fr[kt * 32 + lane];
#pragma unroll
      for (int nt = 0; nt < 2; ++nt)
        o[nt] = WMMA16(pa, v_fr[(kt * 2 + nt) * 32 + lane], o[nt]);
    }
    // store O[bt][l][w][head*32+dh] (f16)
#pragma unroll
    for (int nt = 0; nt < 2; ++nt) {
      const int dh = nt * 16 + ncol;
#pragma unroll
      for (int r = 0; r < 8; ++r) {
        const int lr = mt * 16 + r + 8 * hsel;
        const size_t off =
            ((size_t)(bt * LSEQ + lr) * WDIM + w) * DIMC + head * DH + dh;
        O[off] = (half_t)o[nt][r];
      }
    }
  }
}

// ---------------- Kernel 3: out = attn_out @ Wout + bout ----------------
__global__ __launch_bounds__(256) void out_proj_kernel(
    const half_t* __restrict__ O, const float* __restrict__ Wout,
    const float* __restrict__ bout, float* __restrict__ out) {
  __shared__ v16h a_fr[8 * 32];
  __shared__ v16h b_fr[4 * 32];
  __shared__ float o_lds[64 * 128];  // [j][w]
  half_t* a_h = (half_t*)a_fr;
  half_t* b_h = (half_t*)b_fr;

  const int bx = blockIdx.x;
  const int bt = bx >> 7, l = bx & 127;
  const int j0 = blockIdx.y * 64;
  const int tid = threadIdx.x;
  const int lane = tid & 31, wid = tid >> 5;
  const int wm = wid & 3, wn = wid >> 2;

  v8f acc[2][2] = {};
  const half_t* arow = O + (size_t)(bt * LSEQ + l) * WDIM * DIMC;

  for (int ks = 0; ks < 8; ++ks) {
    const int k0 = ks * 32;
    __syncthreads();
    {  // A tile: f16 source, 2 uint4 per thread, frag-order runs of 8
      const int w = tid >> 1;
      const int kb = (tid & 1) * 16;
      const uint4* src = (const uint4*)(arow + (size_t)w * DIMC + k0 + kb);
#pragma unroll
      for (int h2 = 0; h2 < 2; ++h2) {
        const int kk = kb + h2 * 8;
        *(uint4*)(a_h + (w >> 4) * 512 + a_off(w, kk)) = src[h2];
      }
    }
    {  // B tile: thread owns 8 strided c at fixed j; one packed 16B store
      const int nn = tid & 63;
      const int kg = tid >> 6;  // 0..3
      const float* src = Wout + (size_t)(k0 + kg * 8) * DIMC + j0 + nn;
      Pack8 pk;
#pragma unroll
      for (int u = 0; u < 4; ++u)
        pk.h2[u] = pk2(src[(2 * u) * DIMC], src[(2 * u + 1) * DIMC]);
      *(uint4*)(b_h + (nn >> 4) * 512 + b_off(kg * 8, nn)) = pk.u;
    }
    __syncthreads();
#pragma unroll
    for (int im = 0; im < 2; ++im) {
      const v16h a = a_fr[(wm * 2 + im) * 32 + lane];
#pragma unroll
      for (int in_ = 0; in_ < 2; ++in_) {
        const v16h b = b_fr[(wn * 2 + in_) * 32 + lane];
        acc[im][in_] = WMMA16(a, b, acc[im][in_]);
      }
    }
  }

  __syncthreads();
  // Epilogue: +bout, transpose into o_lds[j][w]
#pragma unroll
  for (int im = 0; im < 2; ++im) {
    const int mt = wm * 2 + im;
#pragma unroll
    for (int in_ = 0; in_ < 2; ++in_) {
      const int nt = wn * 2 + in_;
      const int jl = nt * 16 + (lane & 15);
      const float bj = bout[j0 + jl];
#pragma unroll
      for (int r = 0; r < 8; ++r) {
        const int w = mt * 16 + r + 8 * (lane >> 4);
        o_lds[jl * 128 + w] = acc[im][in_][r] + bj;
      }
    }
  }
  __syncthreads();
  {  // final store: out[bt, c=j0+j, l, :] contiguous along w
    const int j = tid >> 2, q4 = (tid & 3) * 32;
    float* drow = out + ((size_t)(bt * DIMC + j0 + j) * LSEQ + l) * WDIM + q4;
    const float* srow = o_lds + j * 128 + q4;
#pragma unroll
    for (int u = 0; u < 8; ++u)
      ((float4*)drow)[u] = ((const float4*)srow)[u];
  }
}

extern "C" void kernel_launch(void* const* d_in, const int* in_sizes, int n_in,
                              void* d_out, int out_size, void* d_ws,
                              size_t ws_size, hipStream_t stream) {
  const float* x = (const float*)d_in[0];
  const float* rel_bias = (const float*)d_in[1];
  const float* Wqkv = (const float*)d_in[2];
  const float* bqkv = (const float*)d_in[3];
  const float* Wout = (const float*)d_in[4];
  const float* bout = (const float*)d_in[5];
  float* out = (float*)d_out;

  const size_t seg = (size_t)BT * NHEADS * WDIM * LSEQ * DH;  // 33.55M halves
  half_t* Q = (half_t*)d_ws;
  half_t* K = Q + seg;
  half_t* V = K + seg;
  half_t* O = V + seg;  // [bt][l][w][c]

  qkv_proj_kernel<<<dim3(BT * LSEQ, THREE_C / 64), 256, 0, stream>>>(
      x, Wqkv, bqkv, Q, K, V);
  attn_kernel<<<dim3((BT * WDIM * NHEADS) / 2), 64, 0, stream>>>(
      Q, K, V, rel_bias, O);
  out_proj_kernel<<<dim3(BT * LSEQ, DIMC / 64), 256, 0, stream>>>(
      O, Wout, bout, out);
}
